// SSI_DDI_Block_40114994545056
// MI455X (gfx1250) — compile-verified, matrix operand order checked by reading, not run on hardware
//
#include <hip/hip_runtime.h>
#include <hip/hip_bf16.h>
#include <stdint.h>

typedef __attribute__((ext_vector_type(16))) __bf16 v16bf;
typedef __attribute__((ext_vector_type(8)))  __bf16 v8bf;
typedef __attribute__((ext_vector_type(8)))  float  v8f;

#define HIDC 128   // HID == IN_DIM == 128
#define NEG_SLOPE 0.2f
#define NEGINF_ENC 0x007FFFFFu  // encodeF(-inf)

__device__ __forceinline__ unsigned encodeF(float f) {
  unsigned u = __float_as_uint(f);
  return (u & 0x80000000u) ? ~u : (u | 0x80000000u);
}
__device__ __forceinline__ float decodeF(unsigned u) {
  return (u & 0x80000000u) ? __uint_as_float(u & 0x7FFFFFFFu)
                           : __uint_as_float(~u);
}
__device__ __forceinline__ float lrelu(float x) {
  return x > 0.0f ? x : NEG_SLOPE * x;
}
__device__ __forceinline__ void atomAddF(float* p, float v) {
  unsafeAtomicAdd(p, v);   // hardware global_atomic_add_f32
}

// ---------------- init / conversion kernels ----------------
__global__ void fill_f32(float* p, float v, int n) {
  int i = blockIdx.x * blockDim.x + threadIdx.x;
  if (i < n) p[i] = v;
}
__global__ void fill_u32(unsigned* p, unsigned v, int n) {
  int i = blockIdx.x * blockDim.x + threadIdx.x;
  if (i < n) p[i] = v;
}
__global__ void init_out_bias(float* out, const float* __restrict__ bias, int n) {
  int i = blockIdx.x * blockDim.x + threadIdx.x;
  if (i < n) out[i] = bias[i & (HIDC - 1)];
}
// x (row-major f32) -> xbf (row-major bf16)
__global__ void cvt_x_bf16(const float* __restrict__ x, __bf16* __restrict__ xbf, int n) {
  int i = blockIdx.x * blockDim.x + threadIdx.x;
  if (i < n) xbf[i] = (__bf16)x[i];
}
// W [k][c] f32 -> WT [c][k] bf16 (both weights in one pass; 128x128 each)
__global__ void cvt_w_bf16(const float* __restrict__ Wl, const float* __restrict__ Wr,
                           __bf16* __restrict__ WlT, __bf16* __restrict__ WrT) {
  int i = blockIdx.x * blockDim.x + threadIdx.x;   // i < 128*128
  int c = i >> 7, k = i & 127;
  WlT[c * HIDC + k] = (__bf16)Wl[k * HIDC + c];
  WrT[c * HIDC + k] = (__bf16)Wr[k * HIDC + c];
}

__device__ __forceinline__ v16bf cat16(v8bf lo, v8bf hi) {
  return __builtin_shufflevector(lo, hi, 0, 1, 2, 3, 4, 5, 6, 7,
                                 8, 9, 10, 11, 12, 13, 14, 15);
}

// ---------------- GEMM: xl = x@Wl, xr = x@Wr via bf16 WMMA ----------------
// block = 256 (8 waves); wave w handles the 16x16 tile (rowTile=blockIdx.x, colTile=w).
// All operands are pre-converted bf16; per K-chunk each operand is two b128 loads.
__global__ void gat_gemm(const __bf16* __restrict__ xbf,
                         const __bf16* __restrict__ WlT,
                         const __bf16* __restrict__ WrT,
                         float* __restrict__ xl, float* __restrict__ xr, int N) {
  const int lane = threadIdx.x & 31;
  const int wave = threadIdx.x >> 5;         // col tile 0..7
  const int row0 = blockIdx.x * 16;
  const int n0   = wave * 16;
  const int m    = lane & 15;
  const int kb   = (lane >> 4) << 3;         // 0 or 8 (K split across half-waves)
  int row = row0 + m; if (row >= N) row = N - 1;
  const int col = n0 + m;                    // column owned by this lane (B and C/D)

  const __bf16* arow = xbf + (size_t)row * HIDC;
  const __bf16* blc  = WlT + (size_t)col * HIDC;
  const __bf16* brc  = WrT + (size_t)col * HIDC;

  v8f accL = {}; v8f accR = {};
  #pragma unroll
  for (int kc = 0; kc < 4; ++kc) {
    const int k0 = kc * 32;
    // A layout (16-bit, 16x32): lane holds K = kb..kb+7 then 16+kb..16+kb+7
    v16bf a  = cat16(*(const v8bf*)(arow + k0 + kb),
                     *(const v8bf*)(arow + k0 + 16 + kb));
    v16bf bl = cat16(*(const v8bf*)(blc + k0 + kb),
                     *(const v8bf*)(blc + k0 + 16 + kb));
    v16bf br = cat16(*(const v8bf*)(brc + k0 + kb),
                     *(const v8bf*)(brc + k0 + 16 + kb));
    accL = __builtin_amdgcn_wmma_f32_16x16x32_bf16(false, a, false, bl, (short)0, accL, false, false);
    accR = __builtin_amdgcn_wmma_f32_16x16x32_bf16(false, a, false, br, (short)0, accR, false, false);
  }
  // C/D layout: VGPR r -> M = r (lanes 0-15) or r+8 (lanes 16-31); N = lane&15
  const int rbase = row0 + ((lane >> 4) << 3);
  #pragma unroll
  for (int r = 0; r < 8; ++r) {
    const int rr = rbase + r;
    if (rr < N) {
      xl[(size_t)rr * HIDC + col] = accL[r];
      xr[(size_t)rr * HIDC + col] = accR[r];
    }
  }
}

// ---------------- Attention pass 1: logits + segment max ----------------
// one wave per edge (incl. self loops); lane covers 4 channels
__global__ void edge_logits_kernel(const int* __restrict__ ei,
                                   const float* __restrict__ xl,
                                   const float* __restrict__ xr,
                                   const float* __restrict__ att,
                                   float* __restrict__ logits,
                                   unsigned* __restrict__ mEnc,
                                   int E, int Etot) {
  const int e = blockIdx.x * (blockDim.x >> 5) + (threadIdx.x >> 5);
  if (e >= Etot) return;
  const int lane = threadIdx.x & 31;
  int src, dst;
  if (e < E) { src = ei[e]; dst = ei[E + e]; } else { src = dst = e - E; }
  const int c = lane * 4;
  float4 a = *(const float4*)(xl + (size_t)src * HIDC + c);
  float4 b = *(const float4*)(xr + (size_t)dst * HIDC + c);
  float4 w = *(const float4*)(att + c);
  float s = lrelu(a.x + b.x) * w.x + lrelu(a.y + b.y) * w.y +
            lrelu(a.z + b.z) * w.z + lrelu(a.w + b.w) * w.w;
  s += __shfl_xor(s, 1, 32);
  s += __shfl_xor(s, 2, 32);
  s += __shfl_xor(s, 4, 32);            // 8-lane group = one head
  if ((lane & 7) == 0) {
    const int h = lane >> 3;
    logits[(size_t)e * 4 + h] = s;
    atomicMax(mEnc + (size_t)dst * 4 + h, encodeF(s));
  }
}

// ---------------- Attention pass 2: exp + segment sum ----------------
__global__ void edge_exp_kernel(const int* __restrict__ ei,
                                float* __restrict__ logits,
                                const unsigned* __restrict__ mEnc,
                                float* __restrict__ denom,
                                int E, int Etot) {
  const int i = blockIdx.x * blockDim.x + threadIdx.x;
  if (i >= Etot * 4) return;
  const int e = i >> 2, h = i & 3;
  const int dst = (e < E) ? ei[E + e] : (e - E);
  const float mv = decodeF(mEnc[(size_t)dst * 4 + h]);
  const float ev = __expf(logits[i] - mv);
  logits[i] = ev;
  atomAddF(denom + (size_t)dst * 4 + h, ev);
}

// ---------------- Attention pass 3: scatter alpha * xl[src] ----------------
__global__ void edge_scatter_kernel(const int* __restrict__ ei,
                                    const float* __restrict__ xl,
                                    const float* __restrict__ logits,
                                    const float* __restrict__ denom,
                                    float* __restrict__ out,
                                    int E, int Etot) {
  const int e = blockIdx.x * (blockDim.x >> 5) + (threadIdx.x >> 5);
  if (e >= Etot) return;
  const int lane = threadIdx.x & 31;
  int src, dst;
  if (e < E) { src = ei[e]; dst = ei[E + e]; } else { src = dst = e - E; }
  const int h = lane >> 3;
  const float ev = logits[(size_t)e * 4 + h];
  const float dn = denom[(size_t)dst * 4 + h];
  const float alpha = ev / fmaxf(dn, 1e-16f);
  const int c = lane * 4;
  float4 a = *(const float4*)(xl + (size_t)src * HIDC + c);
  float* p = out + (size_t)dst * HIDC + c;
  atomAddF(p + 0, a.x * alpha);
  atomAddF(p + 1, a.y * alpha);
  atomAddF(p + 2, a.z * alpha);
  atomAddF(p + 3, a.w * alpha);
}

// ---------------- SAGPool: scoreAgg[dst] += dot(out[src], W_rel) ----------------
__global__ void score_agg_kernel(const int* __restrict__ ei,
                                 const float* __restrict__ out,
                                 const float* __restrict__ Wrel,
                                 float* __restrict__ scoreAgg, int E) {
  const int e = blockIdx.x * (blockDim.x >> 5) + (threadIdx.x >> 5);
  if (e >= E) return;
  const int lane = threadIdx.x & 31;
  const int src = ei[e], dst = ei[E + e];
  const int c = lane * 4;
  float4 o = *(const float4*)(out + (size_t)src * HIDC + c);
  float4 w = *(const float4*)(Wrel + c);
  float s = o.x * w.x + o.y * w.y + o.z * w.z + o.w * w.w;
  s += __shfl_xor(s, 1, 32);
  s += __shfl_xor(s, 2, 32);
  s += __shfl_xor(s, 4, 32);
  s += __shfl_xor(s, 8, 32);
  s += __shfl_xor(s, 16, 32);
  if (lane == 0) atomAddF(scoreAgg + dst, s);
}

// ---------------- SAGPool: score = agg@W_rel + b + out@W_root; graph max ----
__global__ void score_node_kernel(const float* __restrict__ out,
                                  const float* __restrict__ Wroot,
                                  const float* __restrict__ scoreAgg,
                                  const float* __restrict__ brel,
                                  const int* __restrict__ batch,
                                  float* __restrict__ score,
                                  unsigned* __restrict__ gmEnc, int N) {
  const int n = blockIdx.x * (blockDim.x >> 5) + (threadIdx.x >> 5);
  if (n >= N) return;
  const int lane = threadIdx.x & 31;
  const int c = lane * 4;
  float4 o = *(const float4*)(out + (size_t)n * HIDC + c);
  float4 w = *(const float4*)(Wroot + c);
  float s = o.x * w.x + o.y * w.y + o.z * w.z + o.w * w.w;
  s += __shfl_xor(s, 1, 32);
  s += __shfl_xor(s, 2, 32);
  s += __shfl_xor(s, 4, 32);
  s += __shfl_xor(s, 8, 32);
  s += __shfl_xor(s, 16, 32);
  if (lane == 0) {
    const float sc = scoreAgg[n] + brel[0] + s;
    score[n] = sc;
    atomicMax(gmEnc + batch[n], encodeF(sc));
  }
}

// ---------------- per-graph softmax: exp + sum ----------------
__global__ void graph_exp_kernel(float* __restrict__ score,
                                 const unsigned* __restrict__ gmEnc,
                                 float* __restrict__ gdenom,
                                 const int* __restrict__ batch, int N) {
  const int n = blockIdx.x * blockDim.x + threadIdx.x;
  if (n >= N) return;
  const int b = batch[n];
  const float ev = __expf(score[n] - decodeF(gmEnc[b]));
  score[n] = ev;
  atomAddF(gdenom + b, ev);
}

// ---------------- att_x = out*s ; g = segment_sum(att_x, batch) ----------------
__global__ void final_pool_kernel(const float* __restrict__ out,
                                  const float* __restrict__ score,
                                  const float* __restrict__ gdenom,
                                  const int* __restrict__ batch,
                                  float* __restrict__ g, int N) {
  const int n = blockIdx.x * (blockDim.x >> 5) + (threadIdx.x >> 5);
  if (n >= N) return;
  const int lane = threadIdx.x & 31;
  const int b = batch[n];
  const float s = score[n] / fmaxf(gdenom[b], 1e-16f);
  const int c = lane * 4;
  float4 o = *(const float4*)(out + (size_t)n * HIDC + c);
  float* p = g + (size_t)b * HIDC + c;
  atomAddF(p + 0, o.x * s);
  atomAddF(p + 1, o.y * s);
  atomAddF(p + 2, o.z * s);
  atomAddF(p + 3, o.w * s);
}

extern "C" void kernel_launch(void* const* d_in, const int* in_sizes, int n_in,
                              void* d_out, int out_size, void* d_ws, size_t ws_size,
                              hipStream_t stream) {
  const float* x     = (const float*)d_in[0];
  const int*   ei    = (const int*)  d_in[1];
  const int*   batch = (const int*)  d_in[2];
  const float* Wl    = (const float*)d_in[3];
  const float* Wr    = (const float*)d_in[4];
  const float* att   = (const float*)d_in[5];
  const float* bias  = (const float*)d_in[6];
  const float* Wrel  = (const float*)d_in[7];
  const float* brel  = (const float*)d_in[8];
  const float* Wroot = (const float*)d_in[9];

  const int N    = in_sizes[0] / HIDC;   // 50000
  const int E    = in_sizes[1] / 2;      // 800000
  const int Etot = E + N;                // + self loops
  const int B    = out_size / HIDC - N;  // 512

  float* out = (float*)d_out;            // [N,128]
  float* g   = out + (size_t)N * HIDC;   // [B,128]

  // workspace carve-up
  float*    xl       = (float*)d_ws;                          // N*128
  float*    xr       = xl + (size_t)N * HIDC;                 // N*128
  float*    logits   = xr + (size_t)N * HIDC;                 // Etot*4
  unsigned* mEnc     = (unsigned*)(logits + (size_t)Etot * 4);// N*4
  float*    denom    = (float*)(mEnc + (size_t)N * 4);        // N*4
  float*    scoreAgg = denom + (size_t)N * 4;                 // N
  float*    score    = scoreAgg + N;                          // N
  unsigned* gmEnc    = (unsigned*)(score + N);                // B
  float*    gdenom   = (float*)(gmEnc + B);                   // B
  // bf16 staging (align up to 16 bytes for b128 loads)
  uintptr_t pbf = ((uintptr_t)(gdenom + B) + 15u) & ~(uintptr_t)15u;
  __bf16*   xbf = (__bf16*)pbf;                               // N*128 bf16
  __bf16*   WlT = xbf + (size_t)N * HIDC;                     // 128*128 bf16 (col-major)
  __bf16*   WrT = WlT + HIDC * HIDC;                          // 128*128 bf16 (col-major)

  const int blk = 256;
  const int wpb = blk / 32;   // waves per block

  // init + bf16 staging
  init_out_bias<<<(N * HIDC + blk - 1) / blk, blk, 0, stream>>>(out, bias, N * HIDC);
  fill_f32<<<(B * HIDC + blk - 1) / blk, blk, 0, stream>>>(g, 0.0f, B * HIDC);
  fill_u32<<<(N * 4 + blk - 1) / blk, blk, 0, stream>>>(mEnc, NEGINF_ENC, N * 4);
  fill_f32<<<(N * 4 + blk - 1) / blk, blk, 0, stream>>>(denom, 0.0f, N * 4);
  fill_f32<<<(N + blk - 1) / blk, blk, 0, stream>>>(scoreAgg, 0.0f, N);
  fill_u32<<<(B + blk - 1) / blk, blk, 0, stream>>>(gmEnc, NEGINF_ENC, B);
  fill_f32<<<(B + blk - 1) / blk, blk, 0, stream>>>(gdenom, 0.0f, B);
  cvt_x_bf16<<<(N * HIDC + blk - 1) / blk, blk, 0, stream>>>(x, xbf, N * HIDC);
  cvt_w_bf16<<<(HIDC * HIDC + blk - 1) / blk, blk, 0, stream>>>(Wl, Wr, WlT, WrT);

  // dense transforms via WMMA
  gat_gemm<<<(N + 15) / 16, blk, 0, stream>>>(xbf, WlT, WrT, xl, xr, N);

  // GATv2 attention (3-pass segment softmax + scatter)
  edge_logits_kernel<<<(Etot + wpb - 1) / wpb, blk, 0, stream>>>(ei, xl, xr, att, logits, mEnc, E, Etot);
  edge_exp_kernel<<<(Etot * 4 + blk - 1) / blk, blk, 0, stream>>>(ei, logits, mEnc, denom, E, Etot);
  edge_scatter_kernel<<<(Etot + wpb - 1) / wpb, blk, 0, stream>>>(ei, xl, logits, denom, out, E, Etot);

  // SAGPool score + per-graph softmax + pooled output
  score_agg_kernel<<<(E + wpb - 1) / wpb, blk, 0, stream>>>(ei, out, Wrel, scoreAgg, E);
  score_node_kernel<<<(N + wpb - 1) / wpb, blk, 0, stream>>>(out, Wroot, scoreAgg, brel, batch, score, gmEnc, N);
  graph_exp_kernel<<<(N + blk - 1) / blk, blk, 0, stream>>>(score, gmEnc, gdenom, batch, N);
  final_pool_kernel<<<(N + wpb - 1) / wpb, blk, 0, stream>>>(out, score, gdenom, batch, g, N);
}
